// QNetNode_68848325754965
// MI455X (gfx1250) — matrix operand order repaired and checked by reading, not compiled, not run on hardware
//
#include <hip/hip_runtime.h>
#include <hip/hip_bf16.h>

typedef __attribute__((ext_vector_type(16))) _Float16 v16h;
typedef __attribute__((ext_vector_type(8)))  float    v8f;

#define N_NODES 20000
#define DIM     64
#define BATCH   8
#define EDGES   320000
#define N_ACT   512
#define NI      16
#define NCLS    16

// ---------------------------------------------------------------- utilities
__global__ __launch_bounds__(256) void fill_zero(float* __restrict__ p, long n) {
    long i = (long)blockIdx.x * 256 + threadIdx.x;
    if (i < n) p[i] = 0.0f;
}

// ------------------------------------------------- weight repack (f32->f16)
// Repacks W[K x 64] into WMMA B-fragment order: for K-chunk c, N-tile t,
// lane l holds 16 f16 (32 B) = B[kB+(j&7)+((j&8)?16:0)][t*16 + (l&15)],
// kB = c*32 + ((l&16)?8:0).  Dst flat idx = ((c*4+t)*32 + l)*16 + j.
__global__ __launch_bounds__(256) void pack_w_f16(const float* __restrict__ W,
                                                  _Float16* __restrict__ Wf,
                                                  int K)
{
    int idx = blockIdx.x * 256 + threadIdx.x;     // K*64 total elements
    if (idx >= K * 64) return;
    int j  = idx & 15;
    int l  = (idx >> 4) & 31;
    int t  = (idx >> 9) & 3;
    int c  = idx >> 11;
    int rl = l & 15;
    int kB = c * 32 + ((l & 16) ? 8 : 0);
    Wf[idx] = (_Float16)W[(long)(kB + (j & 7) + ((j & 8) ? 16 : 0)) * DIM + t * 16 + rl];
}

// ---------------------------------------------------------------- WMMA GEMM
// out[M x 64] = relu(X[M x K] @ W[K x 64] + bias (+ add)); preOut = pre-relu.
// One wave per 16-row strip x 64 cols (4 N-tiles); K templated -> full unroll.
// Weights staged once per block into LDS (fragment order), read as v16h
// (2x ds_load_b128). A-fragments: 4x global_load_b128 per chunk.
template <int K>
__global__ __launch_bounds__(256) void wmma_gemm64(
    const float* __restrict__ X, long xbs,
    const _Float16* __restrict__ Wf,
    const float* __restrict__ bias,
    const float* __restrict__ add, long abs_,
    float* __restrict__ out, long obs,
    float* __restrict__ preOut,
    int M)
{
    constexpr int NCHUNK = K / 32;
    __shared__ __attribute__((aligned(32))) _Float16 wlds[K * 64];

    // cooperative stage of all weight fragments (K*128 bytes) into LDS
    {
        const float4* src = (const float4*)Wf;
        float4*       dst = (float4*)wlds;
#pragma unroll
        for (int i = 0; i < K / 32; ++i)               // K*8 float4 / 256 thr
            dst[i * 256 + threadIdx.x] = src[i * 256 + threadIdx.x];
    }
    __syncthreads();

    int wave  = threadIdx.x >> 5;
    int lane  = threadIdx.x & 31;
    int strip = blockIdx.x * 8 + wave;
    if (strip * 16 >= M) return;

    const float* Xb = X + (long)blockIdx.y * xbs;
    int rl = lane & 15;
    int hi = (lane & 16) ? 1 : 0;
    int aRow = strip * 16 + rl;
    const float* xr = Xb + (long)aRow * K;
    int kA = hi ? 8 : 0;
    const v16h* wldsv = (const v16h*)wlds;             // 32B fragments

    v8f acc[4];
#pragma unroll
    for (int t = 0; t < 4; ++t) acc[t] = (v8f){0.f,0.f,0.f,0.f,0.f,0.f,0.f,0.f};

#pragma unroll
    for (int c = 0; c < NCHUNK; ++c) {
        const int kc = c * 32;
        // A tile 16x32 f16: lane holds K runs [kA..kA+7] and [16+kA..16+kA+7]
        float4 p0 = *(const float4*)(xr + kc + kA);
        float4 p1 = *(const float4*)(xr + kc + kA + 4);
        float4 p2 = *(const float4*)(xr + kc + kA + 16);
        float4 p3 = *(const float4*)(xr + kc + kA + 20);
        // B fragments for all 4 N-tiles (ds_load_b128 x2 each)
        v16h b0 = wldsv[(c * 4 + 0) * 32 + lane];
        v16h b1 = wldsv[(c * 4 + 1) * 32 + lane];
        v16h b2 = wldsv[(c * 4 + 2) * 32 + lane];
        v16h b3 = wldsv[(c * 4 + 3) * 32 + lane];
        v16h a;
        a[0]  = (_Float16)p0.x; a[1]  = (_Float16)p0.y;
        a[2]  = (_Float16)p0.z; a[3]  = (_Float16)p0.w;
        a[4]  = (_Float16)p1.x; a[5]  = (_Float16)p1.y;
        a[6]  = (_Float16)p1.z; a[7]  = (_Float16)p1.w;
        a[8]  = (_Float16)p2.x; a[9]  = (_Float16)p2.y;
        a[10] = (_Float16)p2.z; a[11] = (_Float16)p2.w;
        a[12] = (_Float16)p3.x; a[13] = (_Float16)p3.y;
        a[14] = (_Float16)p3.z; a[15] = (_Float16)p3.w;

        acc[0] = __builtin_amdgcn_wmma_f32_16x16x32_f16(
            false, a, false, b0, (short)0, acc[0], false, false);
        acc[1] = __builtin_amdgcn_wmma_f32_16x16x32_f16(
            false, a, false, b1, (short)0, acc[1], false, false);
        acc[2] = __builtin_amdgcn_wmma_f32_16x16x32_f16(
            false, a, false, b2, (short)0, acc[2], false, false);
        acc[3] = __builtin_amdgcn_wmma_f32_16x16x32_f16(
            false, a, false, b3, (short)0, acc[3], false, false);
    }

    // Epilogue: C/D layout VGPR r -> row r (lanes 0-15) / row r+8 (lanes 16-31)
#pragma unroll
    for (int t = 0; t < 4; ++t) {
        int col = t * 16 + rl;
        float bv = bias[col];
#pragma unroll
        for (int r = 0; r < 8; ++r) {
            int row = strip * 16 + r + (hi ? 8 : 0);
            long o = (long)row * DIM + col;
            float v = acc[t][r] + bv;
            if (add)    v += add[(long)blockIdx.y * abs_ + o];
            if (preOut) preOut[(long)blockIdx.y * obs + o] = v;
            out[(long)blockIdx.y * obs + o] = v > 0.f ? v : 0.f;
        }
    }
}

// ------------------------------------------------------- SpMM edge scatter
// pool[b, rows[b,e], :] += edge_w[b,e] * h[b?, cols[b,e], :]
// thread = (b, e, d/4); float4 gather + 4x global_atomic_add_f32 (L2 resident)
__global__ __launch_bounds__(256) void edge_scatter(
    const float* __restrict__ ew, const int* __restrict__ rows,
    const int* __restrict__ cols, const float* __restrict__ hsrc, long hbs,
    float* __restrict__ pool)
{
    long idx = (long)blockIdx.x * 256 + threadIdx.x;   // B*E*16 total, exact
    long per = (long)EDGES * 16;
    int  b   = (int)(idx / per);
    long rem = idx % per;
    int  e   = (int)(rem >> 4);
    int  d   = ((int)(rem & 15)) << 2;
    long be  = (long)b * EDGES + e;
    // stream index/weight arrays ahead (global_prefetch_b8)
    if (be + 8192 < (long)BATCH * EDGES) {
        __builtin_prefetch(rows + be + 8192, 0, 1);
        __builtin_prefetch(cols + be + 8192, 0, 1);
        __builtin_prefetch(ew   + be + 8192, 0, 1);
    }
    int  r   = rows[be];
    int  c   = cols[be];
    float w  = ew[be];
    float4 hv = *(const float4*)(hsrc + (long)b * hbs + (long)c * DIM + d);
    float* dst = pool + ((long)b * N_NODES + (long)r) * DIM + d;
    atomicAdd(dst + 0, w * hv.x);
    atomicAdd(dst + 1, w * hv.y);
    atomicAdd(dst + 2, w * hv.z);
    atomicAdd(dst + 3, w * hv.w);
}

// ------------------------------------------------------- graph mean (sums)
__global__ __launch_bounds__(256) void graph_sum(const float* __restrict__ h,
                                                 float* __restrict__ ge)
{
    int b = blockIdx.y, chunk = blockIdx.x;            // 40 chunks x 500 rows
    int t = threadIdx.x;
    int d = t & 63, ro = t >> 6;
    long base = (long)b * N_NODES * DIM;
    int start = chunk * 500, end = start + 500;
    float s = 0.f;
    for (int r = start + ro; r < end; r += 4) s += h[base + (long)r * DIM + d];
    __shared__ float sm[256];
    sm[t] = s;
    __syncthreads();
    if (t < 64) {
        float v = sm[t] + sm[t + 64] + sm[t + 128] + sm[t + 192];
        atomicAdd(&ge[b * DIM + t], v);
    }
}

// ------------------------------------------------------- label-embed MLP
__global__ __launch_bounds__(64) void label_mlp(
    const float* __restrict__ le1_w, const float* __restrict__ le1_b,
    const float* __restrict__ le2_w, const float* __restrict__ le2_b,
    const int* __restrict__ labels, float* __restrict__ le)
{
    int b = blockIdx.x, t = threadIdx.x;               // 64 threads
    __shared__ float x[64];
    float s = le1_b[t];
    for (int i = 0; i < NI; ++i) {
        int lab = labels[b * NI + i];
        s += le1_w[(long)(i * NCLS + lab) * 64 + t];
    }
    x[t] = s > 0.f ? s : 0.f;
    __syncthreads();
    float dsum = le2_b[t];
    for (int hh = 0; hh < 64; ++hh) dsum += x[hh] * le2_w[hh * 64 + t];
    le[b * 64 + t] = dsum > 0.f ? dsum : 0.f;
}

// ------------------------------------------------------- concat/gather Xe
__global__ __launch_bounds__(256) void concat_feats(
    const float* __restrict__ ge, const float* __restrict__ le,
    const float* __restrict__ h, const int* __restrict__ actions,
    float* __restrict__ Xe)
{
    long idx = (long)blockIdx.x * 256 + threadIdx.x;   // (B*A)*192, exact
    int row = (int)(idx / 192);
    int col = (int)(idx % 192);
    int b = row >> 9, a = row & 511;
    float v;
    if (col < 64)        v = ge[b * 64 + col] * (1.0f / (float)N_NODES);
    else if (col < 128)  v = le[b * 64 + (col - 64)];
    else {
        int node = actions[b * N_ACT + a];
        v = h[((long)b * N_NODES + node) * DIM + (col - 128)];
    }
    Xe[(long)row * 192 + col] = v;
}

// ------------------------------------------------------- final dot + max
__global__ __launch_bounds__(512) void score_max(
    const float* __restrict__ z, const float* __restrict__ linout_w,
    const float* __restrict__ linout_b, float* __restrict__ out)
{
    int b = blockIdx.x, a = threadIdx.x;               // 512 threads
    long row = ((long)b * N_ACT + a) * DIM;
    float s = linout_b[0];
    for (int d = 0; d < DIM; ++d) s += z[row + d] * linout_w[d];
    __shared__ float sm[512];
    sm[a] = s;
    __syncthreads();
    for (int off = 256; off > 0; off >>= 1) {
        if (a < off) sm[a] = fmaxf(sm[a], sm[a + off]);
        __syncthreads();
    }
    if (a == 0) out[b] = sm[0];
}

// =========================================================== launch driver
extern "C" void kernel_launch(void* const* d_in, const int* in_sizes, int n_in,
                              void* d_out, int out_size, void* d_ws, size_t ws_size,
                              hipStream_t stream)
{
    (void)in_sizes; (void)n_in; (void)out_size; (void)ws_size;
    const float* node_features = (const float*)d_in[0];
    const float* w_n2l    = (const float*)d_in[1];
    const float* bias_n2l = (const float*)d_in[2];
    const float* conv_w   = (const float*)d_in[3];
    const float* conv_b   = (const float*)d_in[4];
    const float* lin1_w   = (const float*)d_in[5];
    const float* lin1_b   = (const float*)d_in[6];
    const float* linout_w = (const float*)d_in[7];
    const float* linout_b = (const float*)d_in[8];
    const float* le1_w    = (const float*)d_in[9];
    const float* le1_b    = (const float*)d_in[10];
    const float* le2_w    = (const float*)d_in[11];
    const float* le2_b    = (const float*)d_in[12];
    const float* edge_w   = (const float*)d_in[13];
    const int*   rows     = (const int*)d_in[14];
    const int*   cols     = (const int*)d_in[15];
    const int*   labels   = (const int*)d_in[16];
    const int*   actions  = (const int*)d_in[17];
    float* out = (float*)d_out;

    char* ws = (char*)d_ws;
    float*     im      = (float*)(ws + 0);          // input_message  [20000x64]
    float*     h0      = (float*)(ws + 5120000);    // relu(im)       [20000x64]
    float*     h       = (float*)(ws + 10240000);   // per-batch h    [8x20000x64]
    float*     pool    = (float*)(ws + 51200000);   // n2npool        [8x20000x64]
    float*     ge      = (float*)(ws + 92160000);   // graph sums     [8x64]
    float*     le      = (float*)(ws + 92162048);   // label embed    [8x64]
    float*     Xe      = (float*)(ws + 92164096);   // embed_s_a      [4096x192]
    float*     z       = (float*)(ws + 95309824);   // lin1 out       [4096x64]
    _Float16*  wf_n2l  = (_Float16*)(ws + 96358400); // 128*64 f16 fragments
    _Float16*  wf_conv = (_Float16*)(ws + 96374784); //  64*64 f16 fragments
    _Float16*  wf_lin1 = (_Float16*)(ws + 96382976); // 192*64 f16 fragments

    dim3 blk(256);
    const long NDL = (long)N_NODES * DIM;

    // 0) repack all weight matrices into WMMA B-fragment f16 layout
    pack_w_f16<<<32, blk, 0, stream>>>(w_n2l,  wf_n2l,  128);
    pack_w_f16<<<16, blk, 0, stream>>>(conv_w, wf_conv,  64);
    pack_w_f16<<<48, blk, 0, stream>>>(lin1_w, wf_lin1, 192);

    // 1) input_message + h0 = relu(node_features @ w_n2l + bias)
    wmma_gemm64<128><<<dim3(157, 1), blk, 0, stream>>>(
        node_features, 0, wf_n2l, bias_n2l, nullptr, 0,
        h0, 0, im, N_NODES);

    // 2) two levels of message passing across all 8 graphs
    for (int lv = 0; lv < 2; ++lv) {
        long nPool = (long)BATCH * NDL;
        fill_zero<<<(unsigned)((nPool + 255) / 256), blk, 0, stream>>>(pool, nPool);
        const float* hs = (lv == 0) ? h0 : h;
        long hbs = (lv == 0) ? 0 : NDL;
        long nThr = (long)BATCH * EDGES * 16;
        edge_scatter<<<(unsigned)(nThr / 256), blk, 0, stream>>>(
            edge_w, rows, cols, hs, hbs, pool);
        wmma_gemm64<64><<<dim3(157, BATCH), blk, 0, stream>>>(
            pool, NDL, wf_conv, conv_b, im, 0, h, NDL, nullptr, N_NODES);
    }

    // 3) graph-mean sums
    fill_zero<<<2, blk, 0, stream>>>(ge, BATCH * DIM);
    graph_sum<<<dim3(40, BATCH), blk, 0, stream>>>(h, ge);

    // 4) label embedding MLP
    label_mlp<<<BATCH, 64, 0, stream>>>(le1_w, le1_b, le2_w, le2_b, labels, le);

    // 5) concat [graph_mean | label_embed | action_embed] -> Xe [4096x192]
    concat_feats<<<(BATCH * N_ACT * 192) / 256, blk, 0, stream>>>(
        ge, le, h, actions, Xe);

    // 6) z = relu(Xe @ lin1_w + lin1_b)   [4096x64]
    wmma_gemm64<192><<<dim3(32, 1), blk, 0, stream>>>(
        Xe, 0, wf_lin1, lin1_b, nullptr, 0, z, 0, nullptr, BATCH * N_ACT);

    // 7) raw = z @ linout_w + b; preds = max over actions
    score_max<<<BATCH, 512, 0, stream>>>(z, linout_w, linout_b, out);
}
